// MultiScaleRetention_9002251452482
// MI455X (gfx1250) — matrix operand order repaired and checked by reading, not compiled, not run on hardware
//
#include <hip/hip_runtime.h>
#include <hip/hip_bf16.h>
#include <math.h>

// ---------------------------------------------------------------------------
// CDNA5 (gfx1250) MultiScaleRetention forward.
// bf16 WMMA (V_WMMA_F32_16X16X32_BF16) for all matmuls, fp32 elementwise.
// GEMM A-tiles staged by the Tensor Data Mover (TENSOR_LOAD_TO_LDS/TENSORcnt),
// attention K-tiles by GLOBAL_LOAD_ASYNC_TO_LDS_B128 (ASYNCcnt).
// ---------------------------------------------------------------------------

typedef __attribute__((ext_vector_type(16))) __bf16 bf16x16;
typedef __attribute__((ext_vector_type(8)))  float  floatx8;

union Frag16 {
  bf16x16 v;
  unsigned int u[8];
  unsigned short h[16];
};

// --------------------------- async-to-LDS shim -----------------------------
#if __has_builtin(__builtin_amdgcn_global_load_async_to_lds_b128)
#define USE_ASYNC_LDS 1
#else
#define USE_ASYNC_LDS 0
#endif

typedef int v4i_vs __attribute__((vector_size(16)));
typedef __attribute__((address_space(1))) v4i_vs as1_v4i;  // global int4
typedef __attribute__((address_space(3))) v4i_vs as3_v4i;  // LDS int4

// Copy 16 bytes global -> LDS for this lane (must be 16B aligned both sides).
__device__ __forceinline__ void copy16_g2l(const unsigned short* g, unsigned short* l) {
#if USE_ASYNC_LDS
  __builtin_amdgcn_global_load_async_to_lds_b128(
      (as1_v4i*)(unsigned long long)g,
      (as3_v4i*)(unsigned int)(unsigned long long)l, 0, 0);
#else
  *(uint4*)l = *(const uint4*)g;
#endif
}

__device__ __forceinline__ void wait_async_lds() {
#if USE_ASYNC_LDS
#if __has_builtin(__builtin_amdgcn_s_wait_asynccnt)
  __builtin_amdgcn_s_wait_asynccnt(0);
#else
  __asm__ volatile("s_wait_asynccnt 0" ::: "memory");
#endif
#endif
}

// ------------------------------ TDM shim -----------------------------------
#if __has_builtin(__builtin_amdgcn_tensor_load_to_lds)
#define USE_TDM 1
#else
#define USE_TDM 0
#endif

typedef unsigned int u32x4 __attribute__((ext_vector_type(4)));
typedef int          i32x4 __attribute__((ext_vector_type(4)));
typedef int          i32x8 __attribute__((ext_vector_type(8)));

__device__ __forceinline__ void wait_tensor() {
#if USE_TDM
#if __has_builtin(__builtin_amdgcn_s_wait_tensorcnt)
  __builtin_amdgcn_s_wait_tensorcnt(0);
#else
  __asm__ volatile("s_wait_tensorcnt 0" ::: "memory");
#endif
#endif
}

#if USE_TDM
// 2D tile DMA: rows x tile0 elements (2B each), row stride strideElems, with
// optional LDS padding (padI: 3 -> after every 16 dwords, padA: 3 -> 4 dwords).
__device__ __forceinline__ void tdm_load_2d(const unsigned short* gsrc,
                                            unsigned short* ldst,
                                            unsigned int tile0, unsigned int rows,
                                            unsigned long long strideElems,
                                            unsigned int padI, unsigned int padA,
                                            bool padEn) {
  const unsigned long long ga = (unsigned long long)gsrc;
  const unsigned int laddr = (unsigned int)(unsigned long long)ldst;

  u32x4 g0;
  g0.x = 1u;                                            // count=1, user mode
  g0.y = laddr;                                         // lds_addr
  g0.z = (unsigned int)(ga & 0xFFFFFFFFull);            // global_addr[31:0]
  g0.w = (unsigned int)((ga >> 32) & 0x01FFFFFFull) |   // global_addr[56:32]
         0x80000000u;                                    // type=2 ("image")

  i32x8 g1;
  g1[0] = (int)(0x10000u |                               // data_size=1 (2B)
                (padEn ? (0x100000u | (padI << 22) | (padA << 25)) : 0u));
  g1[1] = (int)((tile0 & 0xFFFFu) << 16);                // tensor_dim0[15:0]
  g1[2] = (int)((tile0 >> 16) | ((rows & 0xFFFFu) << 16));  // td0 hi | td1 lo
  g1[3] = (int)((rows >> 16) | ((tile0 & 0xFFFFu) << 16));  // td1 hi | tile_dim0
  g1[4] = (int)(rows & 0xFFFFu);                         // tile_dim1 (tile_dim2=0)
  g1[5] = (int)(unsigned int)(strideElems & 0xFFFFFFFFull);  // dim0_stride lo
  g1[6] = (int)(unsigned int)((strideElems >> 32) & 0xFFFFull);  // dim0_stride hi
  g1[7] = 0;

  const i32x4 z4 = {0, 0, 0, 0};
#if __clang_major__ >= 23
  const i32x8 z8 = {0, 0, 0, 0, 0, 0, 0, 0};
  __builtin_amdgcn_tensor_load_to_lds(g0, g1, z4, z4, z8, 0);
#else
  __builtin_amdgcn_tensor_load_to_lds(g0, g1, z4, z4, 0);
#endif
}
#endif  // USE_TDM

// ---------------------------------------------------------------------------
__device__ __forceinline__ unsigned short f2bf(float f) {
  unsigned int u = __builtin_bit_cast(unsigned int, f);
  u += 0x7FFFu + ((u >> 16) & 1u);          // round-to-nearest-even
  return (unsigned short)(u >> 16);
}

// A-matrix fragment (16x32 bf16, MxK). Contiguous-k layout (same as frag_b_nk): lane = M row (lane&15),
// half (lane>>4) selects K half; VGPR v holds K = half*16 + v*2 {+0,1}.
__device__ __forceinline__ bf16x16 frag_a(const unsigned short* p, int ld) {
  const int lane = threadIdx.x & 31;
  const int half = lane >> 4;
  const int m = lane & 15;
  Frag16 f;
#pragma unroll
  for (int v = 0; v < 8; ++v) {
    const int k = (half << 4) + (v << 1);
    f.u[v] = *(const unsigned int*)(p + (size_t)m * ld + k);
  }
  return f.v;
}

// B-matrix fragment (32x16 bf16, KxN) from [N][K] (transposed row-major) storage:
// element B[k][n] = p[n*ld + k]. K pairs contiguous -> aligned dword loads.
__device__ __forceinline__ bf16x16 frag_b_nk(const unsigned short* p, int ld) {
  const int lane = threadIdx.x & 31;
  const int kh = (lane >> 4) << 4;
  const int n = lane & 15;
  Frag16 f;
#pragma unroll
  for (int v = 0; v < 8; ++v)
    f.u[v] = *(const unsigned int*)(p + (size_t)n * ld + kh + (v << 1));
  return f.v;
}

// ---------------------------------------------------------------------------
// fp32 -> bf16 conversion (4 elements / thread)
// ---------------------------------------------------------------------------
__global__ void cvt_bf16x4(const float* __restrict__ in,
                           unsigned short* __restrict__ out, size_t n4) {
  const size_t i = (size_t)blockIdx.x * blockDim.x + threadIdx.x;
  if (i >= n4) return;
  const float4 f = ((const float4*)in)[i];
  unsigned short* o = out + i * 4;
  o[0] = f2bf(f.x); o[1] = f2bf(f.y); o[2] = f2bf(f.z); o[3] = f2bf(f.w);
}

// ---------------------------------------------------------------------------
// Generic bf16 GEMM: C(MxN,f32) = A(MxK,bf16 rm) * B(KxN,bf16 rm)
// 256 threads = 8 waves; block tile 128x128, BK=32; per wave 64x32 (4x2 tiles).
// Double-buffered LDS; A staged by TDM (pad_enable reproduces ALD=40),
// B staged transposed [N][K] by the waves.
// ---------------------------------------------------------------------------
#define ALD 40  // 32 data + 8 pad elements per row (TDM: padI=3, padA=3)
#define BLD 34  // even -> dword-aligned fragment reads

__global__ __launch_bounds__(256) void gemm_bf16f32(
    const unsigned short* __restrict__ A, const unsigned short* __restrict__ B,
    float* __restrict__ C, int M, int N, int K) {
  __shared__ __align__(16) unsigned short As[2][128 * ALD];
  __shared__ __align__(16) unsigned short Bs[2][128 * BLD];  // [n][k]
  (void)M;
  const int tid = threadIdx.x;
  const int wid = tid >> 5;
  const int wm = wid >> 2, wn = wid & 3;
  const int bM = blockIdx.y * 128, bN = blockIdx.x * 128;

  // per-thread copy coordinates
  const int arow = tid >> 1;             // A tile row 0..127 (fallback path)
  const int acol = (tid & 1) << 4;       // A tile col base (0 or 16)
  const int bu = tid << 1;               // B: two 8-element chunks per thread

  floatx8 acc[4][2];
#pragma unroll
  for (int a = 0; a < 4; ++a)
#pragma unroll
    for (int b = 0; b < 2; ++b)
#pragma unroll
      for (int v = 0; v < 8; ++v) acc[a][b][v] = 0.f;

  // ---- prologue: stage tile 0 into buffer 0 ----
#if USE_TDM
  if (wid == 0)
    tdm_load_2d(A + (size_t)bM * K, &As[0][0], 32u, 128u,
                (unsigned long long)K, 3u, 3u, true);
#else
#pragma unroll
  for (int j = 0; j < 2; ++j)
    copy16_g2l(A + (size_t)(bM + arow) * K + acol + j * 8,
               &As[0][arow * ALD + acol + j * 8]);
#endif
#pragma unroll
  for (int j = 0; j < 2; ++j) {
    const int u = bu + j;
    const int brow = u >> 4;               // k index 0..31
    const int bcol = (u & 15) << 3;        // n base 0..120
    union { uint4 q; unsigned short h[8]; } t;
    t.q = *(const uint4*)(B + (size_t)brow * N + bN + bcol);
#pragma unroll
    for (int jj = 0; jj < 8; ++jj) Bs[0][(bcol + jj) * BLD + brow] = t.h[jj];
  }
#if USE_TDM
  if (wid == 0) wait_tensor();
#else
  wait_async_lds();
#endif
  __syncthreads();

  for (int k0 = 0; k0 < K; k0 += 32) {
    const int cur = (k0 >> 5) & 1;
    const int nxt = cur ^ 1;
    const bool more = (k0 + 32) < K;

    uint4 breg[2];
    if (more) {
      // kick off TDM staging of the next A tile into the other buffer
#if USE_TDM
      if (wid == 0)
        tdm_load_2d(A + (size_t)bM * K + (k0 + 32), &As[nxt][0], 32u, 128u,
                    (unsigned long long)K, 3u, 3u, true);
#else
#pragma unroll
      for (int j = 0; j < 2; ++j)
        copy16_g2l(A + (size_t)(bM + arow) * K + (k0 + 32) + acol + j * 8,
                   &As[nxt][arow * ALD + acol + j * 8]);
#endif
      // next B tile: global -> registers (stored to LDS after compute)
#pragma unroll
      for (int j = 0; j < 2; ++j) {
        const int u = bu + j;
        breg[j] = *(const uint4*)(B + (size_t)(k0 + 32 + (u >> 4)) * N + bN +
                                  ((u & 15) << 3));
      }
      if (k0 + 64 < K)  // warm L2 for the tile after next
        __builtin_prefetch(B + (size_t)(k0 + 64 + (bu >> 4)) * N + bN +
                               ((bu & 15) << 3), 0, 3);
    }

    // ---- compute from current buffer: 8x v_wmma per wave ----
    bf16x16 bfr[2];
#pragma unroll
    for (int ni = 0; ni < 2; ++ni)
      bfr[ni] = frag_b_nk(&Bs[cur][(wn * 32 + ni * 16) * BLD], BLD);
#pragma unroll
    for (int mi = 0; mi < 4; ++mi) {
      const bf16x16 af = frag_a(&As[cur][(wm * 64 + mi * 16) * ALD], ALD);
#pragma unroll
      for (int ni = 0; ni < 2; ++ni)
        acc[mi][ni] = __builtin_amdgcn_wmma_f32_16x16x32_bf16(
            false, af, false, bfr[ni], (short)0, acc[mi][ni], false, false);
    }

    if (more) {
      // scatter next B tile transposed into the other buffer
#pragma unroll
      for (int j = 0; j < 2; ++j) {
        const int u = bu + j;
        const int brow = u >> 4;
        const int bcol = (u & 15) << 3;
        union { uint4 q; unsigned short h[8]; } t;
        t.q = breg[j];
#pragma unroll
        for (int jj = 0; jj < 8; ++jj)
          Bs[nxt][(bcol + jj) * BLD + brow] = t.h[jj];
      }
    }
#if USE_TDM
    if (wid == 0) wait_tensor();
#else
    wait_async_lds();
#endif
    __syncthreads();
  }

  // Store C: lane = column n, VGPR v -> row v + 8*half
  const int lane = tid & 31;
  const int half = lane >> 4;
  const int n = lane & 15;
#pragma unroll
  for (int mi = 0; mi < 4; ++mi)
#pragma unroll
    for (int ni = 0; ni < 2; ++ni) {
      float* p = C + (size_t)(bM + wm * 64 + mi * 16) * N + bN + wn * 32 + ni * 16 + n;
#pragma unroll
      for (int v = 0; v < 8; ++v)
        p[(size_t)(v + half * 8) * N] = acc[mi][ni][v];
    }
}

// ---------------------------------------------------------------------------
// RoPE (interleaved) + GQA split. qkv fp32 [T][G][6][D] ->
//   qb bf16 [H][T][D] (scaled by 1/sqrt(D)), kb bf16 [G][T][D], vb bf16 [G][T][D]
// ---------------------------------------------------------------------------
__global__ void rope_split(const float* __restrict__ qkv,
                           const float* __restrict__ cosT,
                           const float* __restrict__ sinT,
                           unsigned short* __restrict__ qb,
                           unsigned short* __restrict__ kb,
                           unsigned short* __restrict__ vb, int T) {
  const size_t id = (size_t)blockIdx.x * blockDim.x + threadIdx.x;
  const size_t total = (size_t)T * 4 * 6 * 64;
  if (id >= total) return;
  const int p = (int)(id & 63);
  const int slot = (int)((id >> 6) % 6);
  const int g = (int)((id >> 6) / 6) & 3;
  const int t = (int)((id >> 6) / 24);
  const float* src = qkv + (size_t)t * 3072 + (g * 6 + slot) * 128 + 2 * p;
  const float x1 = src[0], x2 = src[1];
  if (slot < 4) {
    const float c = cosT[t * 64 + p], sn = sinT[t * 64 + p];
    const float scale = 0.088388347648318447f;  // 1/sqrt(128)
    const int hh = g * 4 + slot;
    unsigned short* dst = qb + ((size_t)hh * T + t) * 128 + 2 * p;
    dst[0] = f2bf((x1 * c - x2 * sn) * scale);
    dst[1] = f2bf((x1 * sn + x2 * c) * scale);
  } else if (slot == 4) {
    const float c = cosT[t * 64 + p], sn = sinT[t * 64 + p];
    unsigned short* dst = kb + ((size_t)g * T + t) * 128 + 2 * p;
    dst[0] = f2bf(x1 * c - x2 * sn);
    dst[1] = f2bf(x1 * sn + x2 * c);
  } else {
    unsigned short* dst = vb + ((size_t)g * T + t) * 128 + 2 * p;
    dst[0] = f2bf(x1);
    dst[1] = f2bf(x2);
  }
}

// ---------------------------------------------------------------------------
// Fused retention: per block = (head, 128 query rows). 8 waves x 16 q-rows.
// y = rmsnorm( (q k^T * mask) v ), written fp32 [T][H*D].
// ---------------------------------------------------------------------------
#define VSLD 36  // padded LDS leading dim for transposed V tile

__global__ __launch_bounds__(256) void attn_retention(
    const unsigned short* __restrict__ qb, const unsigned short* __restrict__ kb,
    const unsigned short* __restrict__ vb, const float* __restrict__ mask,
    float* __restrict__ y, int T) {
  __shared__ __align__(16) unsigned short Ks[32 * 128];      // [s][d]
  __shared__ __align__(16) unsigned short Vs[128 * VSLD];    // [d][s] transposed
  __shared__ __align__(16) unsigned short Sbuf[8][16 * 32];  // per-wave S tile

  const int h = blockIdx.y;
  const int g = h >> 2;
  const int qB0 = blockIdx.x * 128;
  const int tid = threadIdx.x, wid = tid >> 5, lane = tid & 31;
  const int qRow0 = qB0 + wid * 16;
  const int half = lane >> 4, n = lane & 15;

  const unsigned short* qh = qb + (size_t)h * T * 128;
  const unsigned short* kh = kb + (size_t)g * T * 128;
  const unsigned short* vh = vb + (size_t)g * T * 128;

  // q fragments for this wave's 16 rows (K = 128 -> 4 chunks of 32)
  bf16x16 qf[4];
#pragma unroll
  for (int kk = 0; kk < 4; ++kk)
    qf[kk] = frag_a(qh + (size_t)qRow0 * 128 + kk * 32, 128);

  floatx8 yacc[8];
#pragma unroll
  for (int i = 0; i < 8; ++i)
#pragma unroll
    for (int v = 0; v < 8; ++v) yacc[i][v] = 0.f;

  const int sEnd = qB0 + 128;  // causal bound for this query block
  for (int s0 = 0; s0 < sEnd; s0 += 32) {
    // cooperative stage: K block [32][128] (async copy-through)
#pragma unroll
    for (int j = 0; j < 2; ++j) {
      const int u = (tid << 1) + j;
      const int s = u >> 4;
      const int d0 = (u & 15) << 3;
      copy16_g2l(kh + (size_t)(s0 + s) * 128 + d0, &Ks[s * 128 + d0]);
    }
    // cooperative stage: V block transposed [128][32]
#pragma unroll
    for (int j = 0; j < 2; ++j) {
      const int u = (tid << 1) + j;
      const int s = u >> 4;
      const int d0 = (u & 15) << 3;
      union { uint4 q; unsigned short hh[8]; } tmp;
      tmp.q = *(const uint4*)(vh + (size_t)(s0 + s) * 128 + d0);
#pragma unroll
      for (int jj = 0; jj < 8; ++jj) Vs[(d0 + jj) * VSLD + s] = tmp.hh[jj];
    }
    wait_async_lds();
    __syncthreads();

    if (s0 <= qRow0 + 15) {  // wave-uniform causal skip
#pragma unroll
      for (int sub = 0; sub < 2; ++sub) {
        floatx8 sc;
#pragma unroll
        for (int v = 0; v < 8; ++v) sc[v] = 0.f;
#pragma unroll
        for (int kk = 0; kk < 4; ++kk)
          sc = __builtin_amdgcn_wmma_f32_16x16x32_bf16(
              false, qf[kk], false,
              frag_b_nk(&Ks[(sub * 16) * 128 + kk * 32], 128),
              (short)0, sc, false, false);
        // decay-mask multiply + pack S tile into per-wave LDS as bf16
        const float* mrow = mask + ((size_t)h * T + qRow0) * T + s0 + sub * 16 + n;
#pragma unroll
        for (int v = 0; v < 8; ++v) {
          const int m = v + half * 8;
          Sbuf[wid][m * 32 + sub * 16 + n] = f2bf(sc[v] * mrow[(size_t)m * T]);
        }
      }
      __asm__ volatile("s_wait_dscnt 0" ::: "memory");  // wave-local LDS RAW
      const bf16x16 sf = frag_a(&Sbuf[wid][0], 32);     // re-laid as A fragment
#pragma unroll
      for (int ni = 0; ni < 8; ++ni)
        yacc[ni] = __builtin_amdgcn_wmma_f32_16x16x32_bf16(
            false, sf, false, frag_b_nk(&Vs[(ni * 16) * VSLD], VSLD),
            (short)0, yacc[ni], false, false);
    }
    __syncthreads();
  }

  // RMS-norm over D=128 entirely in-register: row sums via half-wave shfl_xor
  float rstd[8];
#pragma unroll
  for (int v = 0; v < 8; ++v) {
    float ss = 0.f;
#pragma unroll
    for (int ni = 0; ni < 8; ++ni) ss += yacc[ni][v] * yacc[ni][v];
#pragma unroll
    for (int off = 1; off < 16; off <<= 1) ss += __shfl_xor(ss, off, 32);
    rstd[v] = rsqrtf(ss * (1.0f / 128.0f) + 1e-5f);
  }
#pragma unroll
  for (int ni = 0; ni < 8; ++ni)
#pragma unroll
    for (int v = 0; v < 8; ++v) {
      const int m = v + half * 8;
      y[(size_t)(qRow0 + m) * 2048 + h * 128 + ni * 16 + n] =
          yacc[ni][v] * rstd[v];
    }
}

// ---------------------------------------------------------------------------
// gy = bf16( silu(gpre) * y )
// ---------------------------------------------------------------------------
__global__ void silu_mul(const float* __restrict__ gpre,
                         const float* __restrict__ y,
                         unsigned short* __restrict__ gy, size_t nTot) {
  const size_t i = (size_t)blockIdx.x * blockDim.x + threadIdx.x;
  if (i >= nTot) return;
  const float gv = gpre[i];
  const float s = gv / (1.f + __expf(-gv));
  gy[i] = f2bf(s * y[i]);
}

// ---------------------------------------------------------------------------
extern "C" void kernel_launch(void* const* d_in, const int* in_sizes, int n_in,
                              void* d_out, int out_size, void* d_ws,
                              size_t ws_size, hipStream_t stream) {
  (void)in_sizes; (void)n_in; (void)out_size; (void)ws_size;
  const int T = 2048, C = 2048, H = 16, G = 4, D = 128;

  const float* x    = (const float*)d_in[0];
  const float* cosT = (const float*)d_in[1];
  const float* sinT = (const float*)d_in[2];
  const float* mask = (const float*)d_in[3];
  const float* Wr   = (const float*)d_in[4];
  const float* Wg   = (const float*)d_in[5];
  const float* Wp   = (const float*)d_in[6];

  char* ws = (char*)d_ws;
  size_t off = 0;
  auto carve = [&](size_t bytes) {
    char* p = ws + off;
    off += (bytes + 255) & ~(size_t)255;
    return p;
  };
  unsigned short* xb  = (unsigned short*)carve((size_t)T * C * 2);
  unsigned short* Wrb = (unsigned short*)carve((size_t)C * 3072 * 2);
  unsigned short* Wgb = (unsigned short*)carve((size_t)C * C * 2);
  unsigned short* Wpb = (unsigned short*)carve((size_t)C * C * 2);
  float*          qkv = (float*)carve((size_t)T * 3072 * 4);
  unsigned short* qb  = (unsigned short*)carve((size_t)H * T * D * 2);
  unsigned short* kb  = (unsigned short*)carve((size_t)G * T * D * 2);
  unsigned short* vb  = (unsigned short*)carve((size_t)G * T * D * 2);
  float*          gpre = (float*)carve((size_t)T * C * 4);
  unsigned short* gyb  = (unsigned short*)carve((size_t)T * C * 2);
  float* y = qkv;  // qkv fp32 buffer is dead after rope_split -> reuse for y

  // bf16 conversions
  {
    size_t n4 = (size_t)T * C / 4;
    cvt_bf16x4<<<dim3((unsigned)((n4 + 255) / 256)), 256, 0, stream>>>(x, xb, n4);
    n4 = (size_t)C * 3072 / 4;
    cvt_bf16x4<<<dim3((unsigned)((n4 + 255) / 256)), 256, 0, stream>>>(Wr, Wrb, n4);
    n4 = (size_t)C * C / 4;
    cvt_bf16x4<<<dim3((unsigned)((n4 + 255) / 256)), 256, 0, stream>>>(Wg, Wgb, n4);
    cvt_bf16x4<<<dim3((unsigned)((n4 + 255) / 256)), 256, 0, stream>>>(Wp, Wpb, n4);
  }

  // qkv = x @ W_reten
  gemm_bf16f32<<<dim3(3072 / 128, T / 128), 256, 0, stream>>>(xb, Wrb, qkv, T, 3072, C);

  // RoPE + split
  {
    const size_t total = (size_t)T * G * 6 * 64;
    rope_split<<<dim3((unsigned)((total + 255) / 256)), 256, 0, stream>>>(
        qkv, cosT, sinT, qb, kb, vb, T);
  }

  // gpre = x @ W_gate
  gemm_bf16f32<<<dim3(C / 128, T / 128), 256, 0, stream>>>(xb, Wgb, gpre, T, C, C);

  // retention + rmsnorm -> y
  attn_retention<<<dim3(T / 128, H), 256, 0, stream>>>(qb, kb, vb, mask, y, T);

  // gy = silu(gpre) * y
  {
    const size_t nTot = (size_t)T * C;
    silu_mul<<<dim3((unsigned)((nTot + 255) / 256)), 256, 0, stream>>>(gpre, y, gyb, nTot);
  }

  // out = gy @ W_proj
  gemm_bf16f32<<<dim3(C / 128, T / 128), 256, 0, stream>>>(gyb, Wpb, (float*)d_out, T, C, C);
}
